// Multi_Head_attention_52527450030207
// MI455X (gfx1250) — compile-verified, matrix operand order checked by reading, not run on hardware
//
#include <hip/hip_runtime.h>

// Multi-head attention fwd for MI455X (gfx1250, wave32, WMMA).
// B=4, S=2048, E=1024, H=16, d=64. scale = 1/10, softmax over QUERY axis.
//
// Pipeline:
//   K1 qkv_proj : Y = X @ W^T + b  (f32 -> bf16 operands, f32 accum via
//                 v_wmma_f32_16x16x32_bf16), Q/K/V stored bf16 in workspace.
//   K2 col_stats: per-key m[k]=max_q s, Z[k]=sum_q exp(s-m) (online update).
//   K3 attn_out : out[q,:] = sum_k exp(s(q,k)/10 - m[k])/Z[k] * V[k,:].
//
// Workspace layout (assumes ws_size >= ~52 MB):
//   [0, 16M)   Qbf  (8192 x 1024 bf16)
//   [16M,32M)  Kbf
//   [32M,48M)  Vbf
//   [48M, +512K)  mArr (64 x 2048 f32)
//   [.., +512K)   zArr

typedef __bf16 bf16;
typedef __attribute__((ext_vector_type(16))) __bf16 v16bf;
typedef __attribute__((ext_vector_type(8)))  float  v8f;

#define SEQ 2048
#define EMB 1024
#define NB  4
#define NH  16
#define HD  64

static __device__ inline v8f wmma_bf16(v16bf a, v16bf b, v8f c) {
  // 8 args: (neg_a, A, neg_b, B, c_mod, C, reuse_a, reuse_b)
  return __builtin_amdgcn_wmma_f32_16x16x32_bf16(false, a, false, b,
                                                 (short)0, c, false, false);
}

// ---------------------------------------------------------------------------
// Kernel 1: QKV projection.  grid = (M/128, N/128, 3), block = 256 (8 waves).
// Block tile 128x128, K-step 32.  Wave (wm in 0..3, wn in 0..1) computes a
// 32x64 sub-tile = 2x4 WMMA accumulators.
// ---------------------------------------------------------------------------
__global__ __launch_bounds__(256) void qkv_proj_kernel(
    const float* __restrict__ X,
    const float* __restrict__ Wq, const float* __restrict__ bq,
    const float* __restrict__ Wk, const float* __restrict__ bk,
    const float* __restrict__ Wv, const float* __restrict__ bv,
    bf16* __restrict__ Qbf, bf16* __restrict__ Kbf, bf16* __restrict__ Vbf)
{
  __shared__ bf16 As[128][32];   // 8 KB
  __shared__ bf16 Bs[128][32];   // 8 KB

  const int z = blockIdx.z;
  const float* __restrict__ W    = (z == 0) ? Wq : (z == 1) ? Wk : Wv;
  const float* __restrict__ bias = (z == 0) ? bq : (z == 1) ? bk : bv;
  bf16* __restrict__ Y           = (z == 0) ? Qbf : (z == 1) ? Kbf : Vbf;

  const int m0   = blockIdx.x * 128;
  const int n0   = blockIdx.y * 128;
  const int tid  = threadIdx.x;
  const int lane = tid & 31;
  const int w    = tid >> 5;
  const int wm   = w & 3;   // 32-row slab
  const int wn   = w >> 2;  // 64-col slab
  const int lm   = lane & 15;
  const int lh   = lane >> 4;

  v8f acc[2][4];
#pragma unroll
  for (int i = 0; i < 2; ++i)
#pragma unroll
    for (int j = 0; j < 4; ++j)
#pragma unroll
      for (int r = 0; r < 8; ++r) acc[i][j][r] = 0.0f;

  const int srow = tid >> 1;
  const int skp  = (tid & 1) * 16;

  for (int k0 = 0; k0 < EMB; k0 += 32) {
    // Stage A (x) and B (W) tiles with f32 -> bf16 conversion.
    {
      const float* xp = X + (size_t)(m0 + srow) * EMB + k0 + skp;
      const float* wp = W + (size_t)(n0 + srow) * EMB + k0 + skp;
      v16bf av, bvv;
#pragma unroll
      for (int e = 0; e < 16; ++e) av[e] = (bf16)xp[e];
#pragma unroll
      for (int e = 0; e < 16; ++e) bvv[e] = (bf16)wp[e];
      *(v16bf*)&As[srow][skp] = av;
      *(v16bf*)&Bs[srow][skp] = bvv;
    }
    __syncthreads();

    v16bf af[2], bf[4];
#pragma unroll
    for (int i = 0; i < 2; ++i)
      af[i] = *(const v16bf*)&As[wm * 32 + i * 16 + lm][lh * 16];
#pragma unroll
    for (int j = 0; j < 4; ++j)
      bf[j] = *(const v16bf*)&Bs[wn * 64 + j * 16 + lm][lh * 16];
#pragma unroll
    for (int i = 0; i < 2; ++i)
#pragma unroll
      for (int j = 0; j < 4; ++j)
        acc[i][j] = wmma_bf16(af[i], bf[j], acc[i][j]);
    __syncthreads();
  }

  // Bias + bf16 store.  C layout: lane holds col n = lm, rows r + 8*lh.
#pragma unroll
  for (int j = 0; j < 4; ++j) {
    const int ncol = n0 + wn * 64 + j * 16 + lm;
    const float bb = bias[ncol];
#pragma unroll
    for (int i = 0; i < 2; ++i) {
      const int mbase = m0 + wm * 32 + i * 16 + 8 * lh;
#pragma unroll
      for (int r = 0; r < 8; ++r)
        Y[(size_t)(mbase + r) * EMB + ncol] = (bf16)(acc[i][j][r] + bb);
    }
  }
}

// ---------------------------------------------------------------------------
// Kernel 2: column (query-axis) softmax stats per key.
// grid = (B*H, SEQ/128), block = 256 (8 waves).  Block owns 128 keys; K slab
// staged in LDS; each wave streams 16-query tiles over all 2048 queries with
// online max/expsum, waves merged via LDS at the end.
// ---------------------------------------------------------------------------
__global__ __launch_bounds__(256) void col_stats_kernel(
    const bf16* __restrict__ Qbf, const bf16* __restrict__ Kbf,
    float* __restrict__ mArr, float* __restrict__ zArr)
{
  __shared__ bf16  Ks[128][64];   // 16 KB: 128 keys x 64 dims
  __shared__ float Mw[8][128];    // 4 KB
  __shared__ float Sw[8][128];    // 4 KB

  const int bh = blockIdx.x, b = bh >> 4, h = bh & 15;
  const int kb = blockIdx.y;
  const int tid = threadIdx.x, lane = tid & 31, w = tid >> 5;
  const int lm = lane & 15, lh = lane >> 4;

  // Stage 128 keys x 64 dims of K (bf16) into LDS.
  {
    const int row = tid >> 1, c0 = (tid & 1) * 32;
    const bf16* src =
        Kbf + (size_t)(b * SEQ + kb * 128 + row) * EMB + h * HD + c0;
#pragma unroll
    for (int e = 0; e < 32; ++e) Ks[row][c0 + e] = src[e];
  }
  __syncthreads();

  float M[8], S[8];
#pragma unroll
  for (int t = 0; t < 8; ++t) { M[t] = -3.0e30f; S[t] = 0.0f; }

  for (int i = 0; i < 16; ++i) {
    const int q0 = (w + 8 * i) * 16;
    const bf16* qrow =
        Qbf + (size_t)(b * SEQ + q0 + lm) * EMB + h * HD + lh * 16;
    const v16bf qa0 = *(const v16bf*)qrow;
    const v16bf qa1 = *(const v16bf*)(qrow + 32);

#pragma unroll
    for (int t = 0; t < 8; ++t) {
      const v16bf kf0 = *(const v16bf*)&Ks[t * 16 + lm][lh * 16];
      const v16bf kf1 = *(const v16bf*)&Ks[t * 16 + lm][32 + lh * 16];
      v8f c = {};
      c = wmma_bf16(qa0, kf0, c);
      c = wmma_bf16(qa1, kf1, c);

      float tmax = -3.0e30f;
#pragma unroll
      for (int r = 0; r < 8; ++r) {
        const float s = c[r] * 0.1f;   // fixed 1/10 scale
        c[r] = s;
        tmax = fmaxf(tmax, s);
      }
      tmax = fmaxf(tmax, __shfl_xor(tmax, 16, 32));  // join lane halves
      const float newM = fmaxf(M[t], tmax);
      float tsum = 0.0f;
#pragma unroll
      for (int r = 0; r < 8; ++r) tsum += __expf(c[r] - newM);
      tsum += __shfl_xor(tsum, 16, 32);
      S[t] = S[t] * __expf(M[t] - newM) + tsum;
      M[t] = newM;
    }
  }

  if (lane < 16) {
#pragma unroll
    for (int t = 0; t < 8; ++t) {
      Mw[w][t * 16 + lane] = M[t];
      Sw[w][t * 16 + lane] = S[t];
    }
  }
  __syncthreads();

  if (tid < 128) {
    float gm = -3.0e30f;
#pragma unroll
    for (int ww = 0; ww < 8; ++ww) gm = fmaxf(gm, Mw[ww][tid]);
    float gz = 0.0f;
#pragma unroll
    for (int ww = 0; ww < 8; ++ww) gz += Sw[ww][tid] * __expf(Mw[ww][tid] - gm);
    const size_t idx = (size_t)bh * SEQ + kb * 128 + tid;
    mArr[idx] = gm;
    zArr[idx] = gz;
  }
}

// ---------------------------------------------------------------------------
// Kernel 3: out = P @ V with P[q,k] = exp(s/10 - m[k]) / Z[k].
// grid = (B*H, SEQ/128), block = 256 (8 waves).  Wave owns 16 queries, keeps
// Q fragments + 16x64 f32 accumulators in registers; keys streamed in steps
// of 32 (V staged transposed in LDS; P staged per-wave in LDS to become the
// next WMMA A operand).
// ---------------------------------------------------------------------------
__global__ __launch_bounds__(256) void attn_out_kernel(
    const bf16* __restrict__ Qbf, const bf16* __restrict__ Kbf,
    const bf16* __restrict__ Vbf,
    const float* __restrict__ mArr, const float* __restrict__ zArr,
    float* __restrict__ Out)
{
  __shared__ bf16 Vt[64][32];      // 4 KB : V^T slab  [dim][key]
  __shared__ bf16 Pb[8][16][32];   // 8 KB : per-wave P staging

  const int bh = blockIdx.x, b = bh >> 4, h = bh & 15;
  const int qb = blockIdx.y;
  const int tid = threadIdx.x, lane = tid & 31, w = tid >> 5;
  const int lm = lane & 15, lh = lane >> 4;
  const int q0 = qb * 128 + w * 16;

  // Q fragments for this wave's 16 queries (resident all loop long).
  const bf16* qrow = Qbf + (size_t)(b * SEQ + q0 + lm) * EMB + h * HD + lh * 16;
  const v16bf qa0 = *(const v16bf*)qrow;
  const v16bf qa1 = *(const v16bf*)(qrow + 32);

  v8f acc[4];
#pragma unroll
  for (int dt = 0; dt < 4; ++dt)
#pragma unroll
    for (int r = 0; r < 8; ++r) acc[dt][r] = 0.0f;

  const float* __restrict__ mrow = mArr + (size_t)bh * SEQ;
  const float* __restrict__ zrow = zArr + (size_t)bh * SEQ;

  const int skk = tid & 31;        // staging: key within slab
  const int sd0 = (tid >> 5) * 8;  // staging: 8 dims per thread

  for (int k0 = 0; k0 < SEQ; k0 += 32) {
    __syncthreads();               // prior Vt reads complete
    {
      const bf16* vsrc =
          Vbf + (size_t)(b * SEQ + k0 + skk) * EMB + h * HD + sd0;
      if (k0 + 32 < SEQ)           // hint next slab -> global_prefetch_b8
        __builtin_prefetch(vsrc + 32 * EMB, 0, 1);
#pragma unroll
      for (int e = 0; e < 8; ++e) Vt[sd0 + e][skk] = vsrc[e];
    }
    __syncthreads();

    // Scores for 16 q x 32 k, then P = exp(s/10 - m[k]) / Z[k] in bf16.
#pragma unroll
    for (int t = 0; t < 2; ++t) {
      const int key = k0 + t * 16 + lm;  // lane's key column (same both halves)
      const bf16* krow =
          Kbf + (size_t)(b * SEQ + key) * EMB + h * HD + lh * 16;
      const v16bf kf0 = *(const v16bf*)krow;
      const v16bf kf1 = *(const v16bf*)(krow + 32);
      v8f c = {};
      c = wmma_bf16(qa0, kf0, c);
      c = wmma_bf16(qa1, kf1, c);

      const float mk = mrow[key];
      const float rz = 1.0f / zrow[key];
#pragma unroll
      for (int r = 0; r < 8; ++r) {
        const float p = __expf(c[r] * 0.1f - mk) * rz;
        Pb[w][r + 8 * lh][t * 16 + lm] = (bf16)p;
      }
    }
    asm volatile("s_wait_dscnt 0" ::: "memory");  // P stores visible to loads

    // P (16x32) re-enters WMMA as A; V^T rows are B columns.
    const v16bf pa = *(const v16bf*)&Pb[w][lm][lh * 16];
#pragma unroll
    for (int dt = 0; dt < 4; ++dt) {
      const v16bf vf = *(const v16bf*)&Vt[dt * 16 + lm][lh * 16];
      acc[dt] = wmma_bf16(pa, vf, acc[dt]);
    }
  }

  // Store f32 output in (b, s, h*64 + d) layout.
#pragma unroll
  for (int dt = 0; dt < 4; ++dt) {
    const int dcol = h * HD + dt * 16 + lm;
#pragma unroll
    for (int r = 0; r < 8; ++r)
      Out[(size_t)(b * SEQ + q0 + r + 8 * lh) * EMB + dcol] = acc[dt][r];
  }
}

// ---------------------------------------------------------------------------
extern "C" void kernel_launch(void* const* d_in, const int* in_sizes, int n_in,
                              void* d_out, int out_size, void* d_ws,
                              size_t ws_size, hipStream_t stream) {
  const float* x  = (const float*)d_in[0];
  const float* Wq = (const float*)d_in[1];
  const float* bq = (const float*)d_in[2];
  const float* Wk = (const float*)d_in[3];
  const float* bk = (const float*)d_in[4];
  const float* Wv = (const float*)d_in[5];
  const float* bv = (const float*)d_in[6];
  float* out = (float*)d_out;

  const size_t QKV_BYTES  = (size_t)NB * SEQ * EMB * sizeof(bf16);  // 16 MB
  const size_t STAT_BYTES = (size_t)NB * NH * SEQ * sizeof(float);  // 512 KB
  char* ws = (char*)d_ws;
  bf16*  Qbf  = (bf16*)(ws);
  bf16*  Kbf  = (bf16*)(ws + QKV_BYTES);
  bf16*  Vbf  = (bf16*)(ws + 2 * QKV_BYTES);
  float* mArr = (float*)(ws + 3 * QKV_BYTES);
  float* zArr = (float*)(ws + 3 * QKV_BYTES + STAT_BYTES);

  qkv_proj_kernel<<<dim3(NB * SEQ / 128, EMB / 128, 3), 256, 0, stream>>>(
      x, Wq, bq, Wk, bk, Wv, bv, Qbf, Kbf, Vbf);
  col_stats_kernel<<<dim3(NB * NH, SEQ / 128), 256, 0, stream>>>(
      Qbf, Kbf, mArr, zArr);
  attn_out_kernel<<<dim3(NB * NH, SEQ / 128), 256, 0, stream>>>(
      Qbf, Kbf, Vbf, mArr, zArr, out);
}
